// ImbalanceAwareResidualBlock_75110388073132
// MI455X (gfx1250) — matrix operand order repaired and checked by reading, not compiled, not run on hardware
//
#include <hip/hip_runtime.h>

// ---------------------------------------------------------------------------
// MI455X (gfx1250) implementation.
// Compute-bound (~155 GFLOP vs ~0.2 GB HBM): all large GEMMs + both attention
// matmuls run on v_wmma_f32_16x16x32_bf16 (wave32, f32 accumulate).
// GEMM weight tiles are staged into LDS by the Tensor Data Mover
// (tensor_load_to_lds, TENSORcnt-synchronized, double buffered) instead of the
// legacy VMEM->VGPR->ds_store path. Attention processes 16-query bands with
// the whole 16x2048 score row resident in the 320KB WGP LDS (128KB),
// implementing focal reweighting exactly in log-space
// (p^g = exp(g*(s-m))/Z^g) including the 1e-6 renormalizer.
// ---------------------------------------------------------------------------

typedef unsigned short u16;
typedef unsigned int   u32;
typedef __attribute__((ext_vector_type(16))) __bf16 v16bf;
typedef __attribute__((ext_vector_type(8)))  __bf16 v8bf;
typedef __attribute__((ext_vector_type(8)))  float  v8f;
typedef __attribute__((ext_vector_type(4)))  unsigned int u32x4;
typedef __attribute__((ext_vector_type(4)))  int i32x4;
typedef __attribute__((ext_vector_type(8)))  int i32x8;

__device__ inline u16 f2bfu(float f) {
  u32 u = __builtin_bit_cast(u32, f);
  u32 r = u + 0x7FFFu + ((u >> 16) & 1u);   // round-to-nearest-even
  return (u16)(r >> 16);
}
__device__ inline __bf16 f2bfh(float f) {
  u16 v = f2bfu(f);
  return __builtin_bit_cast(__bf16, v);
}
__device__ inline v8f vzero8() { v8f z = {0.f,0.f,0.f,0.f,0.f,0.f,0.f,0.f}; return z; }

__device__ inline v8f wmma_bf16(v16bf a, v16bf b, v8f c) {
  return __builtin_amdgcn_wmma_f32_16x16x32_bf16(false, a, false, b, (short)0, c,
                                                 false, false);
}

// A-fragment (16x32 bf16): lane (m = lane&15, kh = lane>>4) holds
// elements[0..15] = A[m][kbase + kh*16 + 0..15]
// `p` must already point at A_row + kbase + kh*16.
__device__ inline v16bf ld_a16(const u16* p) {
  v8bf lo = *(const v8bf*)p;
  v8bf hi = *(const v8bf*)(p + 8);
  v16bf r;
#pragma unroll
  for (int i = 0; i < 8; ++i) { r[i] = lo[i]; r[i + 8] = hi[i]; }
  return r;
}

// ---------------------------------------------------------------------------
// TDM: DMA a 64-row x 32-element bf16 tile (row stride = strideElems) from
// global memory into LDS at byte offset ldsByteOff (row-major, contiguous).
// D# per CDNA5 ISA ch.8: group0 = {flags, lds_addr, global_addr, type=2},
// group1 = {data_size=2B, tensor dims == tile dims (tiles always in-bounds),
// tile 32x64, tensor_dim0_stride = K}. Groups 2/3 zero (<=2D tile).
// Wave-level op (EXEC ignored); tracked by TENSORcnt.
// ---------------------------------------------------------------------------
__device__ inline void tdm_load_tile(const u16* gptr, unsigned ldsByteOff,
                                     int strideElems) {
  unsigned long long ga = (unsigned long long)(size_t)gptr;
  u32x4 g0;
  g0[0] = 1u;                                   // count=1 (valid), user mode
  g0[1] = ldsByteOff;                           // lds_addr
  g0[2] = (u32)ga;                              // global_addr[31:0]
  g0[3] = (u32)((ga >> 32) & 0x01FFFFFFu) | (2u << 30);  // addr[56:32] | type=2
  i32x8 g1;
  g1[0] = (int)(1u << 16);                      // data_size=1 -> 2 bytes
  g1[1] = (int)(32u << 16);                     // tensor_dim0 = 32 (lo16)
  g1[2] = (int)(64u << 16);                     // tensor_dim1 = 64 (lo16)
  g1[3] = (int)(32u << 16);                     // tile_dim0 = 32
  g1[4] = (int)64u;                             // tile_dim1 = 64, tile_dim2 = 0
  g1[5] = (int)(u32)strideElems;                // tensor_dim0_stride[31:0]
  g1[6] = 0;                                    // stride hi / dim1_stride lo
  g1[7] = 0;
  i32x4 gz = {0, 0, 0, 0};
#if defined(__clang_major__) && __clang_major__ >= 23
  i32x8 gz8 = {0, 0, 0, 0, 0, 0, 0, 0};
  __builtin_amdgcn_tensor_load_to_lds(g0, g1, gz, gz, gz8, 0);
#else
  __builtin_amdgcn_tensor_load_to_lds(g0, g1, gz, gz, 0);
#endif
}

// ---------------------------------------------------------------------------
// fp32 -> bf16 weight/activation convert
// ---------------------------------------------------------------------------
__global__ __launch_bounds__(256) void cvt_kernel(const float* __restrict__ in,
                                                  u16* __restrict__ out, int n) {
  int i = blockIdx.x * 256 + threadIdx.x;
  if (i < n) out[i] = f2bfu(in[i]);
}

// ---------------------------------------------------------------------------
// Generic WMMA GEMM: out[M][N0] = act(A[M][K](bf16) @ W[N0][K]^T (bf16) + bias)
// Block: 256 threads = 8 waves; tile 128 rows x 64 cols.
// Weight tiles staged by TDM into double-buffered LDS; DMA of tile i+1
// overlaps the WMMAs of tile i. act: 0=none, 1=exact GELU.
// ---------------------------------------------------------------------------
__global__ __launch_bounds__(256)
void gemm_wmma(const u16* __restrict__ A, const u16* __restrict__ W,
               const float* __restrict__ bias, float* __restrict__ outF,
               u16* __restrict__ outB, int M, int K, int N0, int act) {
  __shared__ __align__(64) u16 ldsW[2][64 * 32];   // 2 x 4KB weight tiles
  const int tid = threadIdx.x;
  const int lane = tid & 31, wv = tid >> 5;
  const int colBase = blockIdx.x * 64;
  const int rowBase = blockIdx.y * 128 + wv * 16;
  const int m = lane & 15, kh = lane >> 4;

  v8f acc[4];
#pragma unroll
  for (int t = 0; t < 4; ++t) acc[t] = vzero8();

  const u16* Arow = A + (size_t)(rowBase + m) * K + kh * 16;
  const u16* Wcol = W + (size_t)colBase * K;
  const unsigned lds0 = (unsigned)(size_t)&ldsW[0][0];   // LDS aperture: low 32b
  const unsigned lds1 = (unsigned)(size_t)&ldsW[1][0];

  const int nIt = K >> 5;
  if (wv == 0) tdm_load_tile(Wcol, lds0, K);             // prologue: tile 0

  for (int i = 0; i < nIt; ++i) {
    __syncthreads();   // everyone done reading the buffer the next DMA targets
    if (wv == 0) {
      if (i + 1 < nIt) {
        tdm_load_tile(Wcol + ((i + 1) << 5), (i & 1) ? lds0 : lds1, K);
        __builtin_amdgcn_s_wait_tensorcnt(1);   // tile i landed; tile i+1 in flight
      } else {
        __builtin_amdgcn_s_wait_tensorcnt(0);
      }
    }
    __syncthreads();   // tile i visible to all waves
    const u16* lw = ldsW[i & 1];
    v16bf af = ld_a16(Arow + (i << 5));
#pragma unroll
    for (int t = 0; t < 4; ++t) {
      // B-fragment: lane n=m, elements = 16 contiguous K at kh*16
      v16bf bf = *(const v16bf*)&lw[(t * 16 + m) * 32 + kh * 16];
      acc[t] = wmma_bf16(af, bf, acc[t]);
    }
  }

#pragma unroll
  for (int t = 0; t < 4; ++t) {
    int col = colBase + t * 16 + m;
    float bv = bias[col];
#pragma unroll
    for (int j = 0; j < 8; ++j) {
      int row = rowBase + kh * 8 + j;
      float v = acc[t][j] + bv;
      if (act == 1) v = 0.5f * v * (1.0f + erff(v * 0.70710678118654752f));
      size_t o = (size_t)row * N0 + col;
      if (outF) outF[o] = v;
      if (outB) outB[o] = f2bfu(v);
    }
  }
}

// ---------------------------------------------------------------------------
// LayerNorm over last dim Dw (Dw multiple of 256, <=1024); optional relu;
// writes f32 (ld=Dw) and/or bf16 at (row*ldo + ooff + c). In-place f32 safe.
// ---------------------------------------------------------------------------
__global__ __launch_bounds__(256)
void ln_kernel(const float* __restrict__ in, const float* __restrict__ w,
               const float* __restrict__ b, float* __restrict__ outF,
               u16* __restrict__ outB, int ldo, int ooff, int Dw, int relu) {
  const int row = blockIdx.x, tid = threadIdx.x;
  const float* x = in + (size_t)row * Dw;
  const int per = Dw >> 8;  // <= 4
  float v[4];
  float s = 0.f;
  for (int i = 0; i < per; ++i) { v[i] = x[tid + (i << 8)]; s += v[i]; }
  __shared__ float sred[256];
  __shared__ float smean, svar;
  sred[tid] = s; __syncthreads();
  for (int st = 128; st; st >>= 1) { if (tid < st) sred[tid] += sred[tid + st]; __syncthreads(); }
  if (!tid) smean = sred[0] / (float)Dw;
  __syncthreads();
  float mean = smean;
  s = 0.f;
  for (int i = 0; i < per; ++i) { float d = v[i] - mean; s += d * d; }
  sred[tid] = s; __syncthreads();
  for (int st = 128; st; st >>= 1) { if (tid < st) sred[tid] += sred[tid + st]; __syncthreads(); }
  if (!tid) svar = sred[0] / (float)Dw;
  __syncthreads();
  float rstd = rsqrtf(svar + 1e-5f);
  for (int i = 0; i < per; ++i) {
    int c = tid + (i << 8);
    float y = (v[i] - mean) * rstd * w[c] + b[c];
    if (relu) y = fmaxf(y, 0.f);
    if (outF) outF[(size_t)row * Dw + c] = y;
    if (outB) outB[(size_t)row * ldo + ooff + c] = f2bfu(y);
  }
}

// ---------------------------------------------------------------------------
// Build V^T (bf16): vT[(b*16+h)*64 + d][n] = qkv[b*2048+n][2048 + h*64 + d]
// ---------------------------------------------------------------------------
__global__ __launch_bounds__(256)
void vtrans_kernel(const u16* __restrict__ qkv, u16* __restrict__ vT) {
  int idx = blockIdx.x * 256 + threadIdx.x;   // total 2^22
  int n = idx & 2047;
  int d = (idx >> 11) & 63;
  int bh = idx >> 17;
  int b = bh >> 4, h = bh & 15;
  vT[idx] = qkv[(size_t)(b * 2048 + n) * 3072 + 2048 + h * 64 + d];
}

// ---------------------------------------------------------------------------
// Focal attention, one WG per (b,h,16-row band). 16x2048 scores in LDS.
// ---------------------------------------------------------------------------
__global__ __launch_bounds__(256)
void attn_kernel(const u16* __restrict__ qkv, const u16* __restrict__ vT,
                 u16* __restrict__ attnOut, const float* __restrict__ alphaP,
                 const float* __restrict__ gammaP) {
  __shared__ __align__(64) float S[16 * 2048];   // 128 KB score band
  __shared__ float red1[16 * 16], red2[16 * 16];
  __shared__ float rowM[16], rowZ[16], rowZg[16];

  const int tid = threadIdx.x, lane = tid & 31, wv = tid >> 5;
  const int blk = blockIdx.x;                 // b*2048 + h*128 + rblk
  const int rblk = blk & 127, h = (blk >> 7) & 15, b = blk >> 11;
  const int rowBase = b * 2048 + rblk * 16;
  const int m = lane & 15, kh = lane >> 4;
  const float scale = 0.125f;                 // hd^-0.5, hd=64

  // ---- Phase 1: S = scale * Q Kt  (WMMA) ----
  const u16* qrow = qkv + (size_t)(rowBase + m) * 3072 + h * 64 + kh * 16;
  for (int ct = wv; ct < 128; ct += 8) {
    v8f acc = vzero8();
    const u16* krow =
        qkv + (size_t)(b * 2048 + ct * 16 + m) * 3072 + 1024 + h * 64 + kh * 16;
#pragma unroll
    for (int kk = 0; kk < 64; kk += 32) {
      v16bf af = ld_a16(qrow + kk);
      v16bf bf = *(const v16bf*)(krow + kk);
      acc = wmma_bf16(af, bf, acc);
    }
#pragma unroll
    for (int j = 0; j < 8; ++j)
      S[(kh * 8 + j) * 2048 + ct * 16 + m] = acc[j] * scale;
  }
  __syncthreads();

  // ---- Phase 2: focal-softmax weights, exactly ----
  const float alpha = alphaP[0], gma = gammaP[0];
  const int r = tid >> 4, sc = tid & 15;
  float lmax = -3.4e38f;
  for (int c = sc; c < 2048; c += 16) lmax = fmaxf(lmax, S[r * 2048 + c]);
  red1[r * 16 + sc] = lmax; __syncthreads();
  if (sc == 0) {
    float mm = red1[r * 16];
    for (int i = 1; i < 16; ++i) mm = fmaxf(mm, red1[r * 16 + i]);
    rowM[r] = mm;
  }
  __syncthreads();
  const float mM = rowM[r];
  float z = 0.f, zg = 0.f;
  for (int c = sc; c < 2048; c += 16) {
    float d = S[r * 2048 + c] - mM;
    z += expf(d);
    zg += expf(gma * d);
  }
  red1[r * 16 + sc] = z; red2[r * 16 + sc] = zg; __syncthreads();
  if (sc == 0) {
    float sz = 0.f, szg = 0.f;
    for (int i = 0; i < 16; ++i) { sz += red1[r * 16 + i]; szg += red2[r * 16 + i]; }
    rowZ[r] = sz; rowZg[r] = szg;
  }
  __syncthreads();
  {
    const float Z = rowZ[r], Zg = rowZg[r];
    const float zpg = expf(gma * logf(Z));        // Z^gamma
    const float c1 = alpha / zpg;
    const float denom = c1 * Zg + 1e-6f;          // alpha*sum(p^g) + 1e-6
    const float coef = c1 / denom;
    for (int c = sc; c < 2048; c += 16) {
      int idx = r * 2048 + c;
      S[idx] = expf(gma * (S[idx] - mM)) * coef;  // final attention weight
    }
  }
  __syncthreads();

  // ---- Phase 3: O = P V  (WMMA, K split across 8 waves) ----
  v8f acc3[4];
#pragma unroll
  for (int t = 0; t < 4; ++t) acc3[t] = vzero8();
  const int kBase = wv * 256;
#pragma unroll
  for (int kk = 0; kk < 256; kk += 32) {
    v16bf af;
    const float* sp = &S[m * 2048 + kBase + kk + kh * 16];
#pragma unroll
    for (int i = 0; i < 8; ++i) {
      af[i] = f2bfh(sp[i]);
      af[i + 8] = f2bfh(sp[i + 8]);
    }
#pragma unroll
    for (int t = 0; t < 4; ++t) {
      const u16* vp = vT + (size_t)((b * 16 + h) * 64 + t * 16 + m) * 2048 +
                      kBase + kk + kh * 16;
      v16bf bf = *(const v16bf*)vp;
      acc3[t] = wmma_bf16(af, bf, acc3[t]);
    }
  }
  // partials go into this wave's own (now-consumed) score slab
#pragma unroll
  for (int t = 0; t < 4; ++t)
#pragma unroll
    for (int j = 0; j < 8; ++j)
      S[(kh * 8 + j) * 2048 + kBase + t * 16 + m] = acc3[t][j];
  __syncthreads();

  for (int o = tid; o < 1024; o += 256) {
    int mm = o >> 6, c = o & 63;
    float s = 0.f;
#pragma unroll
    for (int w = 0; w < 8; ++w) s += S[mm * 2048 + w * 256 + c];
    attnOut[(size_t)(rowBase + mm) * 1024 + h * 64 + c] = f2bfu(s);
  }
}

// ---------------------------------------------------------------------------
// Sequence-mean pooling: pools[b][0:1024]=mean(act); if res: pools[b][1024:]=mean(res)
// ---------------------------------------------------------------------------
__global__ __launch_bounds__(256)
void pool_kernel(const float* __restrict__ act, const float* __restrict__ res,
                 float* __restrict__ pools) {
  int b = blockIdx.y;
  int d = blockIdx.x * 256 + threadIdx.x;
  {
    float s = 0.f;
    const float* base = act + (size_t)b * 2048 * 1024 + d;
    for (int n = 0; n < 2048; ++n) s += base[(size_t)n * 1024];
    pools[b * 2048 + d] = s * (1.f / 2048.f);
  }
  if (res) {
    float s = 0.f;
    const float* rb = res + (size_t)b * 2048 * 1024 + d;
    for (int n = 0; n < 2048; ++n) s += rb[(size_t)n * 1024];
    pools[b * 2048 + 1024 + d] = s * (1.f / 2048.f);
  }
}

// ---------------------------------------------------------------------------
// AdaptiveResidualGate: sigmoid(W2 relu(LN(W1 p + b1)) + b2), one WG per batch
// ---------------------------------------------------------------------------
__global__ __launch_bounds__(256)
void gate_kernel(const float* __restrict__ pools, const float* __restrict__ w1,
                 const float* __restrict__ b1, const float* __restrict__ lnw,
                 const float* __restrict__ lnb, const float* __restrict__ w2,
                 const float* __restrict__ b2, float* __restrict__ gate) {
  const int b = blockIdx.x, tid = threadIdx.x;
  const float* p = pools + b * 2048;
  const float* wr = w1 + (size_t)tid * 2048;
  float acc = b1[tid];
  for (int i = 0; i < 2048; ++i) acc += p[i] * wr[i];
  __shared__ float sred[256];
  __shared__ float hv[256];
  __shared__ float smean, svar;
  sred[tid] = acc; __syncthreads();
  for (int st = 128; st; st >>= 1) { if (tid < st) sred[tid] += sred[tid + st]; __syncthreads(); }
  if (!tid) smean = sred[0] * (1.f / 256.f);
  __syncthreads();
  float d = acc - smean;
  sred[tid] = d * d; __syncthreads();
  for (int st = 128; st; st >>= 1) { if (tid < st) sred[tid] += sred[tid + st]; __syncthreads(); }
  if (!tid) svar = sred[0] * (1.f / 256.f);
  __syncthreads();
  float hval = d * rsqrtf(svar + 1e-5f) * lnw[tid] + lnb[tid];
  hv[tid] = fmaxf(hval, 0.f);
  __syncthreads();
  for (int kk = 0; kk < 4; ++kk) {
    int o = tid + (kk << 8);
    const float* w2r = w2 + (size_t)o * 256;
    float a2 = b2[o];
    for (int i = 0; i < 256; ++i) a2 += hv[i] * w2r[i];
    gate[b * 1024 + o] = 1.f / (1.f + expf(-a2));
  }
}

// ---------------------------------------------------------------------------
// xout = xin + act * gate[b][d]   (gate broadcast over sequence)
// ---------------------------------------------------------------------------
__global__ __launch_bounds__(256)
void addgate_kernel(const float* __restrict__ xin, const float* __restrict__ act,
                    const float* __restrict__ gate, float* __restrict__ xout,
                    int nTot) {
  int idx = blockIdx.x * 256 + threadIdx.x;
  if (idx < nTot) {
    int d = idx & 1023;
    int b = idx >> 21;   // 2048*1024 = 2^21 per batch
    xout[idx] = xin[idx] + act[idx] * gate[b * 1024 + d];
  }
}

// ---------------------------------------------------------------------------
// Final: cw = softmax(ch @ cgw2^T + cgb2); out = x2*cw0 + residual*cw1
// ---------------------------------------------------------------------------
__global__ __launch_bounds__(256)
void final_blend_kernel(const float* __restrict__ ch, const float* __restrict__ cgw2,
                        const float* __restrict__ cgb2, const float* __restrict__ x2,
                        const float* __restrict__ residual, float* __restrict__ out) {
  const int row = blockIdx.x, tid = threadIdx.x;
  __shared__ float s0[256], s1[256];
  __shared__ float w0s;
  float p0 = 0.f, p1 = 0.f;
  const float* cr = ch + (size_t)row * 1024;
  for (int c = tid; c < 1024; c += 256) {
    float v = cr[c];
    p0 += v * cgw2[c];
    p1 += v * cgw2[1024 + c];
  }
  s0[tid] = p0; s1[tid] = p1; __syncthreads();
  for (int st = 128; st; st >>= 1) {
    if (tid < st) { s0[tid] += s0[tid + st]; s1[tid] += s1[tid + st]; }
    __syncthreads();
  }
  if (!tid) {
    float d0 = s0[0] + cgb2[0], d1 = s1[0] + cgb2[1];
    float mx = fmaxf(d0, d1);
    float e0 = expf(d0 - mx), e1 = expf(d1 - mx);
    w0s = e0 / (e0 + e1);
  }
  __syncthreads();
  float w0 = w0s, w1 = 1.f - w0;
  const float* xr = x2 + (size_t)row * 1024;
  const float* rr = residual + (size_t)row * 1024;
  float* orow = out + (size_t)row * 1024;
  for (int c = tid; c < 1024; c += 256) orow[c] = xr[c] * w0 + rr[c] * w1;
}

// ---------------------------------------------------------------------------
extern "C" void kernel_launch(void* const* d_in, const int* in_sizes, int n_in,
                              void* d_out, int out_size, void* d_ws, size_t ws_size,
                              hipStream_t stream) {
  (void)in_sizes; (void)n_in; (void)out_size; (void)ws_size;
  const float* x        = (const float*)d_in[0];
  const float* residual = (const float*)d_in[1];
  const float* n1w = (const float*)d_in[2];  const float* n1b = (const float*)d_in[3];
  const float* n2w = (const float*)d_in[4];  const float* n2b = (const float*)d_in[5];
  const float* n3w = (const float*)d_in[6];  const float* n3b = (const float*)d_in[7];
  const float* qkv_w = (const float*)d_in[8];  const float* qkv_b = (const float*)d_in[9];
  const float* proj_w = (const float*)d_in[10]; const float* proj_b = (const float*)d_in[11];
  const float* f_alpha = (const float*)d_in[12]; const float* f_gamma = (const float*)d_in[13];
  const float* m_w1 = (const float*)d_in[14]; const float* m_b1 = (const float*)d_in[15];
  const float* m_w2 = (const float*)d_in[16]; const float* m_b2 = (const float*)d_in[17];
  const float* g1_w1 = (const float*)d_in[18]; const float* g1_b1 = (const float*)d_in[19];
  const float* g1_lnw = (const float*)d_in[20]; const float* g1_lnb = (const float*)d_in[21];
  const float* g1_w2 = (const float*)d_in[22]; const float* g1_b2 = (const float*)d_in[23];
  const float* g2_w1 = (const float*)d_in[24]; const float* g2_b1 = (const float*)d_in[25];
  const float* g2_lnw = (const float*)d_in[26]; const float* g2_lnb = (const float*)d_in[27];
  const float* g2_w2 = (const float*)d_in[28]; const float* g2_b2 = (const float*)d_in[29];
  const float* cg_w1 = (const float*)d_in[30]; const float* cg_b1 = (const float*)d_in[31];
  const float* cg_lnw = (const float*)d_in[32]; const float* cg_lnb = (const float*)d_in[33];
  const float* cg_w2 = (const float*)d_in[34]; const float* cg_b2 = (const float*)d_in[35];
  float* out = (float*)d_out;

  const int M = 4096;          // B*N = 2*2048
  char* ws = (char*)d_ws;
  size_t off = 0;
  auto take = [&](size_t bytes) -> char* {
    char* p = ws + off;
    off += (bytes + 255) & ~(size_t)255;
    return p;
  };
  u16*   Wqkv  = (u16*)take((size_t)3072 * 1024 * 2);
  u16*   Wproj = (u16*)take((size_t)1024 * 1024 * 2);
  u16*   Wm1   = (u16*)take((size_t)4096 * 1024 * 2);
  u16*   Wm2   = (u16*)take((size_t)1024 * 4096 * 2);
  u16*   Wcg   = (u16*)take((size_t)1024 * 2048 * 2);
  u16*   NX    = (u16*)take((size_t)M * 1024 * 2);    // ln out / attn out (bf16)
  u16*   ARENA = (u16*)take((size_t)M * 4096 * 2);    // qkv (24MB) then mlp hidden (32MB)
  u16*   VT    = (u16*)take((size_t)32 * 64 * 2048 * 2);
  float* F32A  = (float*)take((size_t)M * 1024 * 4);  // attnProj -> mlp_out -> ch
  float* XBUF  = (float*)take((size_t)M * 1024 * 4);  // x1 then x2 (in-place)
  u16*   COMB  = (u16*)take((size_t)M * 2048 * 2);    // [ln(x2) | ln(res)] bf16
  float* POOLS = (float*)take((size_t)2 * 2048 * 4);
  float* GATE  = (float*)take((size_t)2 * 1024 * 4);

  // 1) weights -> bf16
  cvt_kernel<<<(3072 * 1024 + 255) / 256, 256, 0, stream>>>(qkv_w, Wqkv, 3072 * 1024);
  cvt_kernel<<<(1024 * 1024 + 255) / 256, 256, 0, stream>>>(proj_w, Wproj, 1024 * 1024);
  cvt_kernel<<<(4096 * 1024 + 255) / 256, 256, 0, stream>>>(m_w1, Wm1, 4096 * 1024);
  cvt_kernel<<<(4096 * 1024 + 255) / 256, 256, 0, stream>>>(m_w2, Wm2, 4096 * 1024);
  cvt_kernel<<<(2048 * 1024 + 255) / 256, 256, 0, stream>>>(cg_w1, Wcg, 2048 * 1024);

  // 2) nx = LN(x, norm1) -> bf16
  ln_kernel<<<M, 256, 0, stream>>>(x, n1w, n1b, (float*)nullptr, NX, 1024, 0, 1024, 0);

  // 3) qkv = nx @ qkv_w^T + qkv_b  (bf16 out)
  gemm_wmma<<<dim3(3072 / 64, M / 128), 256, 0, stream>>>(
      NX, Wqkv, qkv_b, (float*)nullptr, ARENA, M, 1024, 3072, 0);

  // 4) V^T
  vtrans_kernel<<<(32 * 64 * 2048) / 256, 256, 0, stream>>>(ARENA, VT);

  // 5) focal attention -> NX (bf16, overwrites ln1 output)
  attn_kernel<<<4096, 256, 0, stream>>>(ARENA, VT, NX, f_alpha, f_gamma);

  // 6) attnProj = attn @ proj_w^T + proj_b  (f32)
  gemm_wmma<<<dim3(1024 / 64, M / 128), 256, 0, stream>>>(
      NX, Wproj, proj_b, F32A, (u16*)nullptr, M, 1024, 1024, 0);

  // 7) pools = [mean(attnProj), mean(residual)]
  pool_kernel<<<dim3(4, 2), 256, 0, stream>>>(F32A, residual, POOLS);
  // 8) gate1
  gate_kernel<<<2, 256, 0, stream>>>(POOLS, g1_w1, g1_b1, g1_lnw, g1_lnb, g1_w2, g1_b2, GATE);
  // 9) x1 = x + attnProj * gate1
  addgate_kernel<<<(M * 1024) / 256, 256, 0, stream>>>(x, F32A, GATE, XBUF, M * 1024);

  // 10) nx = LN(x1, norm2) -> bf16
  ln_kernel<<<M, 256, 0, stream>>>(XBUF, n2w, n2b, (float*)nullptr, NX, 1024, 0, 1024, 0);

  // 11) h = GELU(nx @ mlp_w1^T + b1) -> bf16 (ARENA, qkv dead)
  gemm_wmma<<<dim3(4096 / 64, M / 128), 256, 0, stream>>>(
      NX, Wm1, m_b1, (float*)nullptr, ARENA, M, 1024, 4096, 1);

  // 12) mlp_out = h @ mlp_w2^T + b2 -> f32 (F32A, attnProj dead)
  gemm_wmma<<<dim3(1024 / 64, M / 128), 256, 0, stream>>>(
      ARENA, Wm2, m_b2, F32A, (u16*)nullptr, M, 4096, 1024, 0);

  // 13) refresh activation half of pools (residual half kept)
  pool_kernel<<<dim3(4, 2), 256, 0, stream>>>(F32A, (const float*)nullptr, POOLS);
  // 14) gate2
  gate_kernel<<<2, 256, 0, stream>>>(POOLS, g2_w1, g2_b1, g2_lnw, g2_lnb, g2_w2, g2_b2, GATE);
  // 15) x2 = x1 + mlp_out * gate2 (in place)
  addgate_kernel<<<(M * 1024) / 256, 256, 0, stream>>>(XBUF, F32A, GATE, XBUF, M * 1024);

  // 16/17) combined = [LN(x2,n3) | LN(residual,n3)] bf16
  ln_kernel<<<M, 256, 0, stream>>>(XBUF, n3w, n3b, (float*)nullptr, COMB, 2048, 0, 1024, 0);
  ln_kernel<<<M, 256, 0, stream>>>(residual, n3w, n3b, (float*)nullptr, COMB, 2048, 1024, 1024, 0);

  // 18) chPre = combined @ cg_w1^T + cg_b1 -> f32 (F32A, mlp_out dead)
  gemm_wmma<<<dim3(1024 / 64, M / 128), 256, 0, stream>>>(
      COMB, Wcg, cg_b1, F32A, (u16*)nullptr, M, 2048, 1024, 0);

  // 19) ch = relu(LN(chPre, cg_ln)) in place
  ln_kernel<<<M, 256, 0, stream>>>(F32A, cg_lnw, cg_lnb, F32A, (u16*)nullptr, 1024, 0, 1024, 1);

  // 20) softmax cross-gate blend -> out
  final_blend_kernel<<<M, 256, 0, stream>>>(F32A, cg_w2, cg_b2, XBUF, residual, out);
}
